// AudioQuantizer_40003325395701
// MI455X (gfx1250) — matrix operand order repaired and checked by reading, not compiled, not run on hardware
//
#include <hip/hip_runtime.h>
#include <hip/hip_bf16.h>

typedef __attribute__((ext_vector_type(16))) _Float16 v16h;
typedef __attribute__((ext_vector_type(2)))  _Float16 v2h;
typedef __attribute__((ext_vector_type(8)))  float    v8f;

#define N_FRAMES 32768
#define K_TOK    8192
#define DMODEL   512

// ---------------- kernel 1: codebook fp32 -> f16 + row squared norms --------
__global__ __launch_bounds__(256) void cvt_codebook(const float* __restrict__ cb,
                                                    _Float16* __restrict__ cb16,
                                                    float* __restrict__ csq) {
  const int row  = blockIdx.x * 8 + (threadIdx.x >> 5);
  const int lane = threadIdx.x & 31;
  const float* src = cb   + (size_t)row * DMODEL;
  _Float16*    dst = cb16 + (size_t)row * DMODEL;
  float acc = 0.0f;
#pragma unroll
  for (int i = 0; i < 8; ++i) {
    const int c = i * 64 + lane * 2;
    const float2 f = *(const float2*)(src + c);
    acc += f.x * f.x + f.y * f.y;
    v2h h; h.x = (_Float16)f.x; h.y = (_Float16)f.y;
    *(v2h*)(dst + c) = h;
  }
#pragma unroll
  for (int m = 16; m >= 1; m >>= 1) acc += __shfl_xor(acc, m, 32);
  if (lane == 0) csq[row] = acc;
}

// ---------------- kernel 2: WMMA GEMM + fused argmin ------------------------
// Wave owns 16 rows of x (A resident in 128 VGPRs as f16 fragments).
// Per 32-entry codebook tile: 32 x v_wmma_f32_16x16x32_f16, then score/min.
__global__ __launch_bounds__(256, 1) void vq_argmin(const float* __restrict__ x,
                                                    const _Float16* __restrict__ cb16,
                                                    const float* __restrict__ csq,
                                                    int* __restrict__ idx) {
  const int lane    = threadIdx.x & 31;
  const int wave    = threadIdx.x >> 5;
  const int nl      = lane & 15;  // A: row-within-tile, B/C: column-within-tile
  const int h       = lane >> 4;  // lane half
  const int rowbase = blockIdx.x * 128 + wave * 16;

  // Load & convert this wave's 16x512 slice of x into register A fragments.
  // A order used here (16-bit, contiguous k, same as B): lanes 0-15 row m=lane, v0..7 = K 0..15;
  // lanes 16-31 same rows, v0..7 = K 16..31.
  v16h a[16];
  {
    const float* xr = x + (size_t)(rowbase + nl) * DMODEL;
#pragma unroll
    for (int s = 0; s < 16; ++s) {
      const int d0 = s * 32 + h * 16;
      const v8f lo = *(const v8f*)(xr + d0);
      const v8f hi = *(const v8f*)(xr + d0 + 8);
      v16h t;
#pragma unroll
      for (int e = 0; e < 8; ++e) { t[e] = (_Float16)lo[e]; t[8 + e] = (_Float16)hi[e]; }
      a[s] = t;
    }
  }

  float minv[8];
  int   mini[8];
#pragma unroll
  for (int j = 0; j < 8; ++j) { minv[j] = 3.0e38f; mini[j] = 0; }

  for (int k0 = 0; k0 < K_TOK; k0 += 32) {
    // B layout (32x16): lane holds column n = lane&15 (= codebook row k0+nl),
    // v0..7 pack K = (lane>>4)*16 .. +15 -> 16 contiguous f16 from that row.
    const _Float16* b0p = cb16 + (size_t)(k0 + nl) * DMODEL + h * 16;
    const _Float16* b1p = b0p + (size_t)16 * DMODEL;
    v8f c0 = {};
    v8f c1 = {};
#pragma unroll
    for (int s = 0; s < 16; ++s) {
      const v16h b0 = *(const v16h*)(b0p + s * 32);
      const v16h b1 = *(const v16h*)(b1p + s * 32);
      c0 = __builtin_amdgcn_wmma_f32_16x16x32_f16(false, a[s], false, b0,
                                                  (short)0, c0, false, false);
      c1 = __builtin_amdgcn_wmma_f32_16x16x32_f16(false, a[s], false, b1,
                                                  (short)0, c1, false, false);
    }
    // score = ||c||^2 - 2 * <x, c>  (||x||^2 dropped: constant per row)
    const float q0 = csq[k0 + nl];
    const float q1 = csq[k0 + 16 + nl];
#pragma unroll
    for (int j = 0; j < 8; ++j) {
      const float s0 = q0 - 2.0f * c0[j];
      if (s0 < minv[j]) { minv[j] = s0; mini[j] = k0 + nl; }
      const float s1 = q1 - 2.0f * c1[j];
      if (s1 < minv[j]) { minv[j] = s1; mini[j] = k0 + 16 + nl; }
    }
  }

  // C layout: lane l holds column n=l&15, rows m = j + 8*(l>>4). Reduce the
  // running (min,idx) across the 16 columns of each lane half.
#pragma unroll
  for (int j = 0; j < 8; ++j) {
#pragma unroll
    for (int m = 1; m <= 8; m <<= 1) {
      const float ov = __shfl_xor(minv[j], m, 32);
      const int   oi = __shfl_xor(mini[j], m, 32);
      if (ov < minv[j] || (ov == minv[j] && oi < mini[j])) {
        minv[j] = ov; mini[j] = oi;
      }
    }
  }
  if (nl == 0) {
#pragma unroll
    for (int j = 0; j < 8; ++j) idx[rowbase + h * 8 + j] = mini[j];
  }
}

// ---------------- kernel 3: embedding gather (bandwidth-bound) --------------
__global__ __launch_bounds__(128) void gather_rows(const float* __restrict__ emb,
                                                   const int* __restrict__ idx,
                                                   float* __restrict__ out) {
  const int r = blockIdx.x;
  const int k = idx[r];
  const float4* src = (const float4*)(emb + (size_t)k * DMODEL);
  float4*       dst = (float4*)(out + (size_t)r * DMODEL);
  dst[threadIdx.x] = src[threadIdx.x];
}

extern "C" void kernel_launch(void* const* d_in, const int* in_sizes, int n_in,
                              void* d_out, int out_size, void* d_ws, size_t ws_size,
                              hipStream_t stream) {
  const float* x   = (const float*)d_in[0];  // [32768, 512] fp32
  const float* cb  = (const float*)d_in[1];  // [8192, 512] fp32
  const float* emb = (const float*)d_in[2];  // [8192, 512] fp32
  float* out = (float*)d_out;                // [32768, 512] fp32

  char* ws = (char*)d_ws;
  _Float16* cb16 = (_Float16*)ws;                                   // 8 MiB
  float*    csq  = (float*)(ws + (size_t)K_TOK * DMODEL * 2);       // 32 KiB
  int*      idx  = (int*)(ws + (size_t)K_TOK * DMODEL * 2 + (size_t)K_TOK * 4);

  cvt_codebook<<<K_TOK / 8, 256, 0, stream>>>(cb, cb16, csq);
  vq_argmin<<<N_FRAMES / 128, 256, 0, stream>>>(x, cb16, csq, idx);
  gather_rows<<<N_FRAMES, 128, 0, stream>>>(emb, idx, out);
}
